// DCP_5257039970653
// MI455X (gfx1250) — matrix operand order repaired and checked by reading, not compiled, not run on hardware
//
#include <hip/hip_runtime.h>
#include <math.h>

// ---------------- problem constants ----------------
#define NCLOUD 32      // 16 batches x {src,tgt}
#define NPTS   1024
#define KNBR   20
#define BATCH  16
#define SEQ    1024
#define DM     512
#define NH     4
#define DHEAD  128
#define DFF2   1024

typedef __attribute__((ext_vector_type(16))) __bf16 bf16x16;
typedef __attribute__((ext_vector_type(8)))  float  v8f;

__device__ __forceinline__ unsigned short f2bf(float f) {
  unsigned int u = __float_as_uint(f);
  u += 0x7FFFu + ((u >> 16) & 1u);           // round-to-nearest-even
  return (unsigned short)(u >> 16);
}

// ---------------- generic batched WMMA GEMM ----------------
// C[z] = act( (A[z] @ op(B[z])) * scale[n] + bias[n] )
// op(B) = B (K x N, ldb) or B^T (transB: B stored N x K, ldb)
// per-operand z offset:  base + (z / zDiv) * s1 + (z % zDiv) * s2
struct GemmP {
  const float* A; const float* Bw; float* C;
  const float* bias; const float* scale;
  long long sA1, sA2, sB1, sB2, sC1, sC2;
  int zDivA, zDivB, zDivC;
  int lda, ldb, ldc;
  int M, N, K;
  int transB, relu;
};

#define BM 128
#define BN 64
#define BK 32

// Double-buffered, software-pipelined WMMA GEMM.
// FAST: K%32==0, lda%4==0, ldb%4==0, M%128==0 (all launches except DGCNN L1).
template <bool FAST>
__global__ __launch_bounds__(256)
void k_gemm_wmma(GemmP p) {
  // LDS fragments pre-swizzled into WMMA operand layout (bf16), ping-pong
  __shared__ __align__(32) unsigned short sA[2][8 * 32 * 16]; // 128x32 tile
  __shared__ __align__(32) unsigned short sB[2][4 * 32 * 16]; // 32x64 tile

  const int tid  = threadIdx.x;
  const int lane = tid & 31;
  const int wave = tid >> 5;
  const int wm   = wave & 3;   // 4 wave-rows  (32 rows each)
  const int wn   = wave >> 2;  // 2 wave-cols  (32 cols each)
  const int z    = blockIdx.z;
  const int m0   = blockIdx.y * BM;
  const int n0   = blockIdx.x * BN;

  const float* __restrict__ A  = p.A  + (long long)(z / p.zDivA) * p.sA1 + (long long)(z % p.zDivA) * p.sA2;
  const float* __restrict__ Bm = p.Bw + (long long)(z / p.zDivB) * p.sB1 + (long long)(z % p.zDivB) * p.sB2;
  float*       __restrict__ C  = p.C  + (long long)(z / p.zDivC) * p.sC1 + (long long)(z % p.zDivC) * p.sC2;

  v8f acc[2][2];
  const v8f vzero = {0.f,0.f,0.f,0.f,0.f,0.f,0.f,0.f};
  acc[0][0]=vzero; acc[0][1]=vzero; acc[1][0]=vzero; acc[1][1]=vzero;

  float4 ra[4];   // staged A: 4 float4/thread (128x32 tile)
  float4 rb[2];   // staged B: 2 float4/thread (32x64 tile)

  // ---- tile load: global -> registers (no LDS traffic) ----
  auto loadTile = [&](int kt) {
    #pragma unroll
    for (int q = 0; q < 4; ++q) {
      int lin = (tid << 2) + q;
      int row = lin >> 3;            // 8 float4 per 32-float row
      int k4  = (lin & 7) << 2;
      if (FAST) {
        const float* src = A + (long long)(m0 + row) * p.lda + (kt + k4);
        ra[q] = *(const float4*)src;
        if (q == 0 && kt + 2 * BK < p.K) __builtin_prefetch(src + 2 * BK, 0, 1);
      } else {
        float v0=0.f, v1=0.f, v2=0.f, v3=0.f;
        int gm = m0 + row;
        if (gm < p.M) {
          const float* src = A + (long long)gm * p.lda + (kt + k4);
          if (kt + k4 + 0 < p.K) v0 = src[0];
          if (kt + k4 + 1 < p.K) v1 = src[1];
          if (kt + k4 + 2 < p.K) v2 = src[2];
          if (kt + k4 + 3 < p.K) v3 = src[3];
        }
        ra[q] = make_float4(v0, v1, v2, v3);
      }
    }
    #pragma unroll
    for (int q = 0; q < 2; ++q) {
      int lin = (tid << 1) + q;
      if (!p.transB) {
        int kk = lin >> 4;           // 16 float4 per 64-float k-row
        int n4 = (lin & 15) << 2;
        int gk = kt + kk;
        if (FAST) {
          rb[q] = *(const float4*)(Bm + (long long)gk * p.ldb + (n0 + n4));
        } else {
          if (gk < p.K) {
            const float* s = Bm + (long long)gk * p.ldb + (n0 + n4);
            rb[q] = make_float4(s[0], s[1], s[2], s[3]);
          } else {
            rb[q] = make_float4(0.f, 0.f, 0.f, 0.f);
          }
        }
      } else {                       // B[k][n] = Bsrc[n][k] (contiguous in k)
        int n  = lin >> 3;           // 8 float4 per n (32 k)
        int k4 = (lin & 7) << 2;
        if (FAST) {
          rb[q] = *(const float4*)(Bm + (long long)(n0 + n) * p.ldb + (kt + k4));
        } else {
          const float* s = Bm + (long long)(n0 + n) * p.ldb + (kt + k4);
          float v0=0.f, v1=0.f, v2=0.f, v3=0.f;
          if (kt + k4 + 0 < p.K) v0 = s[0];
          if (kt + k4 + 1 < p.K) v1 = s[1];
          if (kt + k4 + 2 < p.K) v2 = s[2];
          if (kt + k4 + 3 < p.K) v3 = s[3];
          rb[q] = make_float4(v0, v1, v2, v3);
        }
      }
    }
  };

  // ---- tile store: registers -> swizzled bf16 fragments in LDS ----
  auto storeTile = [&](int pb) {
    #pragma unroll
    for (int q = 0; q < 4; ++q) {
      int lin = (tid << 2) + q;
      int row = lin >> 3;
      int k4  = (lin & 7) << 2;
      int r = row >> 4, mm = row & 15;
      int half = k4 >> 4;                          // A frag lane halves: k>=16 -> lanes 16..31
      int e0   = k4 & 15;                          // 4 consecutive elements
      unsigned int lo = (unsigned int)f2bf(ra[q].x) | ((unsigned int)f2bf(ra[q].y) << 16);
      unsigned int hi = (unsigned int)f2bf(ra[q].z) | ((unsigned int)f2bf(ra[q].w) << 16);
      unsigned int* d = (unsigned int*)&sA[pb][(((r << 5) + (half << 4) + mm) << 4) + e0];
      d[0] = lo; d[1] = hi;
    }
    #pragma unroll
    for (int q = 0; q < 2; ++q) {
      int lin = (tid << 1) + q;
      if (!p.transB) {
        int kk = lin >> 4;
        int n4 = (lin & 15) << 2;
        int c = n4 >> 4;
        int lanebase = (kk >> 4) << 4;             // k>=16 -> lanes 16..31
        int e = kk & 15;
        float vv[4] = {rb[q].x, rb[q].y, rb[q].z, rb[q].w};
        #pragma unroll
        for (int u = 0; u < 4; ++u) {
          int nn = (n4 + u) & 15;
          sB[pb][(((c << 5) + lanebase + nn) << 4) + e] = f2bf(vv[u]);
        }
      } else {
        int n  = lin >> 3;
        int k4 = (lin & 7) << 2;
        int c = n >> 4, nn = n & 15;
        int lanebase = (k4 >> 4) << 4;
        int e0 = k4 & 15;                          // 4 consecutive elements
        unsigned int lo = (unsigned int)f2bf(rb[q].x) | ((unsigned int)f2bf(rb[q].y) << 16);
        unsigned int hi = (unsigned int)f2bf(rb[q].z) | ((unsigned int)f2bf(rb[q].w) << 16);
        unsigned int* d = (unsigned int*)&sB[pb][(((c << 5) + lanebase + nn) << 4) + e0];
        d[0] = lo; d[1] = hi;
      }
    }
  };

  // ---- compute: 2x2 WMMA per wave from fragment-order LDS ----
  auto compTile = [&](int pb) {
    #pragma unroll
    for (int i = 0; i < 2; ++i) {
      int r = (wm << 1) + i;
      bf16x16 aF = *(const bf16x16*)&sA[pb][((r << 5) + lane) << 4];
      #pragma unroll
      for (int j = 0; j < 2; ++j) {
        int c = (wn << 1) + j;
        bf16x16 bF = *(const bf16x16*)&sB[pb][((c << 5) + lane) << 4];
        acc[i][j] = __builtin_amdgcn_wmma_f32_16x16x32_bf16(
            false, aF, false, bF, (short)0, acc[i][j], false, false);
      }
    }
  };

  // ---- software pipeline: load(t+1) overlaps wmma(t); 1 barrier / tile ----
  const int nt = (p.K + BK - 1) >> 5;
  loadTile(0);
  int pb = 0;
  for (int t = 0; t < nt; ++t) {
    storeTile(pb);
    __syncthreads();
    if (t + 1 < nt) loadTile((t + 1) << 5);
    compTile(pb);
    pb ^= 1;
  }

  // ---- epilogue: C/D layout -> row = vgpr + (lane>>4)*8, col = lane&15 ----
  const int colIn  = lane & 15;
  const int rowAdd = (lane >> 4) << 3;
  #pragma unroll
  for (int i = 0; i < 2; ++i) {
    #pragma unroll
    for (int j = 0; j < 2; ++j) {
      int colBase = n0 + (wn << 5) + (j << 4) + colIn;
      float sc = p.scale ? p.scale[colBase] : 1.0f;
      float bs = p.bias  ? p.bias [colBase] : 0.0f;
      #pragma unroll
      for (int v = 0; v < 8; ++v) {
        int row = m0 + (wm << 5) + (i << 4) + rowAdd + v;
        float y = acc[i][j][v] * sc + bs;
        if (p.relu) y = fmaxf(y, 0.0f);
        C[(long long)row * p.ldc + colBase] = y;
      }
    }
  }
}

// ---------------- small support kernels ----------------
__global__ void k_bnfold(const float* g, const float* b, const float* m,
                         const float* v, float* scale, float* shift, int C) {
  int i = blockIdx.x * 64 + threadIdx.x;
  if (i < C) {
    float s = g[i] * rsqrtf(v[i] + 1e-5f);
    scale[i] = s;
    shift[i] = b[i] - m[i] * s;
  }
}

__global__ void k_topk(const float* X, int* idx) {
  int t = blockIdx.x * blockDim.x + threadIdx.x;
  if (t >= NCLOUD * NPTS) return;
  int c = t / NPTS, i = t % NPTS;
  const float* xc = X + (long long)c * NPTS * 3;
  float xi0 = xc[i*3], xi1 = xc[i*3+1], xi2 = xc[i*3+2];
  float bv[KNBR]; int bi[KNBR];
  #pragma unroll
  for (int q = 0; q < KNBR; ++q) { bv[q] = -3.4e38f; bi[q] = 0; }
  for (int j = 0; j < NPTS; ++j) {
    float d0 = xc[j*3]-xi0, d1 = xc[j*3+1]-xi1, d2 = xc[j*3+2]-xi2;
    float d = -(d0*d0 + d1*d1 + d2*d2);
    if (d > bv[KNBR-1]) {
      int q = KNBR - 1;
      while (q > 0 && bv[q-1] < d) { bv[q] = bv[q-1]; bi[q] = bi[q-1]; --q; }
      bv[q] = d; bi[q] = j;
    }
  }
  for (int q = 0; q < KNBR; ++q) idx[(long long)t * KNBR + q] = bi[q];
}

__global__ void k_h0(const float* X, const int* idx, float* h) {
  long long t = (long long)blockIdx.x * blockDim.x + threadIdx.x;
  if (t >= (long long)NCLOUD * NPTS * KNBR) return;
  int kk = (int)(t % KNBR);
  long long pn = t / KNBR;
  int n = (int)(pn % NPTS);
  long long c = pn / NPTS;
  const float* xc = X + c * NPTS * 3;
  int j = idx[pn * KNBR + kk];
  float* o = h + (pn * KNBR + kk) * 6;
  o[0]=xc[j*3]; o[1]=xc[j*3+1]; o[2]=xc[j*3+2];
  o[3]=xc[n*3]; o[4]=xc[n*3+1]; o[5]=xc[n*3+2];
}

__global__ void k_maxk(const float* h, float* feat, int C, int coff) {
  long long t = (long long)blockIdx.x * blockDim.x + threadIdx.x;
  if (t >= (long long)NCLOUD * NPTS * C) return;
  int ch = (int)(t % C);
  long long pn = t / C;
  const float* src = h + (pn * KNBR) * C + ch;
  float m = src[0];
  #pragma unroll 4
  for (int kk = 1; kk < KNBR; ++kk) m = fmaxf(m, src[(long long)kk * C]);
  feat[pn * 512 + coff + ch] = m;
}

__global__ __launch_bounds__(256) void k_softmax(float* a, float scale) {
  const int L = 1024;
  float* p = a + (long long)blockIdx.x * L;
  __shared__ float red[8];
  int tid = threadIdx.x, lane = tid & 31, wv = tid >> 5;
  float v[4]; float mx = -3.4e38f;
  #pragma unroll
  for (int i = 0; i < 4; ++i) { v[i] = p[tid + i*256] * scale; mx = fmaxf(mx, v[i]); }
  for (int o = 16; o; o >>= 1) mx = fmaxf(mx, __shfl_xor(mx, o));
  if (lane == 0) red[wv] = mx;
  __syncthreads();
  if (tid == 0) { float t = red[0]; for (int i = 1; i < 8; ++i) t = fmaxf(t, red[i]); red[0] = t; }
  __syncthreads();
  mx = red[0];
  __syncthreads();
  float s = 0.f;
  #pragma unroll
  for (int i = 0; i < 4; ++i) { v[i] = __expf(v[i] - mx); s += v[i]; }
  for (int o = 16; o; o >>= 1) s += __shfl_xor(s, o);
  if (lane == 0) red[wv] = s;
  __syncthreads();
  if (tid == 0) { float t = 0.f; for (int i = 0; i < 8; ++i) t += red[i]; red[0] = t; }
  __syncthreads();
  float inv = 1.0f / red[0];
  #pragma unroll
  for (int i = 0; i < 4; ++i) p[tid + i*256] = v[i] * inv;
}

__global__ __launch_bounds__(256)
void k_add_ln(const float* x, const float* r, const float* g, const float* bb, float* out) {
  const int D = 512;
  long long base = (long long)blockIdx.x * D;
  int tid = threadIdx.x, lane = tid & 31, wv = tid >> 5;
  __shared__ float red[8]; __shared__ float sMean, sRstd;
  float a0 = x[base + tid]       + r[base + tid];
  float a1 = x[base + tid + 256] + r[base + tid + 256];
  float s = a0 + a1;
  for (int o = 16; o; o >>= 1) s += __shfl_xor(s, o);
  if (lane == 0) red[wv] = s;
  __syncthreads();
  if (tid == 0) { float t = 0.f; for (int i = 0; i < 8; ++i) t += red[i]; sMean = t / 512.f; }
  __syncthreads();
  float m = sMean;
  float d0 = a0 - m, d1 = a1 - m;
  float q = d0*d0 + d1*d1;
  for (int o = 16; o; o >>= 1) q += __shfl_xor(q, o);
  __syncthreads();
  if (lane == 0) red[wv] = q;
  __syncthreads();
  if (tid == 0) { float t = 0.f; for (int i = 0; i < 8; ++i) t += red[i]; sRstd = rsqrtf(t / 512.f + 1e-6f); }
  __syncthreads();
  float rs = sRstd;
  out[base + tid]       = g[tid]       * d0 * rs + bb[tid];
  out[base + tid + 256] = g[tid + 256] * d1 * rs + bb[tid + 256];
}

__global__ void k_add(const float* a, const float* b, float* o, long long n) {
  long long i = (long long)blockIdx.x * 256 + threadIdx.x;
  if (i < n) o[i] = a[i] + b[i];
}

__global__ __launch_bounds__(256)
void k_matched(const float* ptr_, const float* X, float* matched) {
  int b = blockIdx.x >> 10, n = blockIdx.x & 1023;
  const float* prow = ptr_ + ((long long)b * SEQ + n) * SEQ;
  const float* tgt  = X + (long long)(b * 2 + 1) * NPTS * 3;
  float a0 = 0.f, a1 = 0.f, a2 = 0.f;
  for (int j = threadIdx.x; j < SEQ; j += 256) {
    float w = prow[j];
    a0 += w * tgt[j*3]; a1 += w * tgt[j*3+1]; a2 += w * tgt[j*3+2];
  }
  __shared__ float red[8];
  float vv[3] = {a0, a1, a2};
  for (int c = 0; c < 3; ++c) {
    float v = vv[c];
    for (int o = 16; o; o >>= 1) v += __shfl_xor(v, o);
    if ((threadIdx.x & 31) == 0) red[threadIdx.x >> 5] = v;
    __syncthreads();
    if (threadIdx.x == 0) {
      float t = 0.f; for (int i = 0; i < 8; ++i) t += red[i];
      matched[((long long)b * NPTS + n) * 3 + c] = t;
    }
    __syncthreads();
  }
}

__global__ __launch_bounds__(256)
void k_centroid(const float* X, const float* matched, float* cents) {
  int b = blockIdx.x;
  const float* src = X + (long long)(b * 2) * NPTS * 3;
  const float* mt  = matched + (long long)b * NPTS * 3;
  float s[6] = {0,0,0,0,0,0};
  for (int j = threadIdx.x; j < NPTS; j += 256) {
    s[0] += src[j*3]; s[1] += src[j*3+1]; s[2] += src[j*3+2];
    s[3] += mt[j*3];  s[4] += mt[j*3+1];  s[5] += mt[j*3+2];
  }
  __shared__ float red[8];
  for (int c = 0; c < 6; ++c) {
    float v = s[c];
    for (int o = 16; o; o >>= 1) v += __shfl_xor(v, o);
    if ((threadIdx.x & 31) == 0) red[threadIdx.x >> 5] = v;
    __syncthreads();
    if (threadIdx.x == 0) { float t = 0.f; for (int i = 0; i < 8; ++i) t += red[i]; cents[b*6 + c] = t / (float)NPTS; }
    __syncthreads();
  }
}

__global__ __launch_bounds__(256)
void k_hmat(const float* X, const float* matched, const float* cents, float* hm) {
  int b = blockIdx.x;
  const float* src = X + (long long)(b * 2) * NPTS * 3;
  const float* mt  = matched + (long long)b * NPTS * 3;
  const float* ce  = cents + b * 6;
  float sc0 = ce[0], sc1 = ce[1], sc2 = ce[2], mc0 = ce[3], mc1 = ce[4], mc2 = ce[5];
  float acc[9] = {0,0,0,0,0,0,0,0,0};
  for (int j = threadIdx.x; j < NPTS; j += 256) {
    float s0 = src[j*3]-sc0, s1 = src[j*3+1]-sc1, s2 = src[j*3+2]-sc2;
    float m0 = mt[j*3]-mc0,  m1 = mt[j*3+1]-mc1,  m2 = mt[j*3+2]-mc2;
    acc[0]+=s0*m0; acc[1]+=s0*m1; acc[2]+=s0*m2;
    acc[3]+=s1*m0; acc[4]+=s1*m1; acc[5]+=s1*m2;
    acc[6]+=s2*m0; acc[7]+=s2*m1; acc[8]+=s2*m2;
  }
  __shared__ float red[8];
  for (int c = 0; c < 9; ++c) {
    float v = acc[c];
    for (int o = 16; o; o >>= 1) v += __shfl_xor(v, o);
    if ((threadIdx.x & 31) == 0) red[threadIdx.x >> 5] = v;
    __syncthreads();
    if (threadIdx.x == 0) { float t = 0.f; for (int i = 0; i < 8; ++i) t += red[i]; hm[b*9 + c] = t; }
    __syncthreads();
  }
}

__global__ void k_svd(const float* Hm, const float* cents, float* out) {
  int b = threadIdx.x;
  if (b >= BATCH) return;
  float A[3][3], Ks[3][3], V[3][3];
  for (int r = 0; r < 3; ++r) for (int c = 0; c < 3; ++c) A[r][c] = Hm[b*9 + r*3 + c];
  for (int i = 0; i < 3; ++i)
    for (int j = 0; j < 3; ++j) {
      float s = 0.f; for (int k = 0; k < 3; ++k) s += A[k][i] * A[k][j];
      Ks[i][j] = s;
    }
  for (int r = 0; r < 3; ++r) for (int c = 0; c < 3; ++c) V[r][c] = (r == c) ? 1.f : 0.f;
  const int PQ[3][2] = {{0,1},{0,2},{1,2}};
  for (int sweep = 0; sweep < 12; ++sweep) {
    for (int e = 0; e < 3; ++e) {
      int p = PQ[e][0], q = PQ[e][1];
      float apq = Ks[p][q];
      if (fabsf(apq) < 1e-20f) continue;
      float theta = (Ks[q][q] - Ks[p][p]) / (2.f * apq);
      float tt = (theta >= 0.f ? 1.f : -1.f) / (fabsf(theta) + sqrtf(1.f + theta*theta));
      float cc = rsqrtf(1.f + tt*tt);
      float ss = tt * cc;
      for (int k = 0; k < 3; ++k) {
        float kp = Ks[k][p], kq = Ks[k][q];
        Ks[k][p] = cc*kp - ss*kq; Ks[k][q] = ss*kp + cc*kq;
      }
      for (int k = 0; k < 3; ++k) {
        float kp = Ks[p][k], kq = Ks[q][k];
        Ks[p][k] = cc*kp - ss*kq; Ks[q][k] = ss*kp + cc*kq;
      }
      for (int k = 0; k < 3; ++k) {
        float vp = V[k][p], vq = V[k][q];
        V[k][p] = cc*vp - ss*vq; V[k][q] = ss*vp + cc*vq;
      }
    }
  }
  float w[3] = {Ks[0][0], Ks[1][1], Ks[2][2]};
  for (int i = 0; i < 2; ++i)
    for (int j = i + 1; j < 3; ++j)
      if (w[j] > w[i]) {
        float tw = w[i]; w[i] = w[j]; w[j] = tw;
        for (int k = 0; k < 3; ++k) { float tv = V[k][i]; V[k][i] = V[k][j]; V[k][j] = tv; }
      }
  float U[3][3];
  for (int i = 0; i < 3; ++i) {
    float si = sqrtf(fmaxf(w[i], 0.f));
    float inv = 1.f / fmaxf(si, 1e-12f);
    for (int r = 0; r < 3; ++r)
      U[r][i] = (A[r][0]*V[0][i] + A[r][1]*V[1][i] + A[r][2]*V[2][i]) * inv;
  }
  float R[3][3];
  for (int i = 0; i < 3; ++i)
    for (int j = 0; j < 3; ++j)
      R[i][j] = V[i][0]*U[j][0] + V[i][1]*U[j][1] + V[i][2]*U[j][2];
  float det = R[0][0]*(R[1][1]*R[2][2]-R[1][2]*R[2][1])
            - R[0][1]*(R[1][0]*R[2][2]-R[1][2]*R[2][0])
            + R[0][2]*(R[1][0]*R[2][1]-R[1][1]*R[2][0]);
  if (det < 0.f) { R[2][0] = -R[2][0]; R[2][1] = -R[2][1]; R[2][2] = -R[2][2]; }
  const float* sc = cents + b*6;
  const float* mc = sc + 3;
  float tv[3];
  for (int i = 0; i < 3; ++i)
    tv[i] = mc[i] - (R[i][0]*sc[0] + R[i][1]*sc[1] + R[i][2]*sc[2]);
  for (int r = 0; r < 3; ++r) for (int c = 0; c < 3; ++c) out[b*12 + r*3 + c] = R[r][c];
  for (int c = 0; c < 3; ++c) out[b*12 + 9 + c] = tv[c];
}

// ---------------- input leaf indices ----------------
// Assumption: d_in[0] = x; params flattened in jax tree order (sorted dict keys):
// dgcnn{bn1..bn5{b,g,m,v}, w1..w5}, transformer{dec{ffn{d1{b,w},d2{b,w}},
// ln1{b,g},ln2,ln3, mha1{wk{b,w},wo,wq,wv}, mha2}, enc{ffn,ln1,ln2,mha}}
enum {
  I_X = 0,
  I_BN1_B, I_BN1_G, I_BN1_M, I_BN1_V,
  I_BN2_B, I_BN2_G, I_BN2_M, I_BN2_V,
  I_BN3_B, I_BN3_G, I_BN3_M, I_BN3_V,
  I_BN4_B, I_BN4_G, I_BN4_M, I_BN4_V,
  I_BN5_B, I_BN5_G, I_BN5_M, I_BN5_V,
  I_W1, I_W2, I_W3, I_W4, I_W5,
  I_DEC_FFN_D1_B, I_DEC_FFN_D1_W, I_DEC_FFN_D2_B, I_DEC_FFN_D2_W,
  I_DEC_LN1_B, I_DEC_LN1_G, I_DEC_LN2_B, I_DEC_LN2_G, I_DEC_LN3_B, I_DEC_LN3_G,
  I_DEC_MHA1_WK_B, I_DEC_MHA1_WK_W, I_DEC_MHA1_WO_B, I_DEC_MHA1_WO_W,
  I_DEC_MHA1_WQ_B, I_DEC_MHA1_WQ_W, I_DEC_MHA1_WV_B, I_DEC_MHA1_WV_W,
  I_DEC_MHA2_WK_B, I_DEC_MHA2_WK_W, I_DEC_MHA2_WO_B, I_DEC_MHA2_WO_W,
  I_DEC_MHA2_WQ_B, I_DEC_MHA2_WQ_W, I_DEC_MHA2_WV_B, I_DEC_MHA2_WV_W,
  I_ENC_FFN_D1_B, I_ENC_FFN_D1_W, I_ENC_FFN_D2_B, I_ENC_FFN_D2_W,
  I_ENC_LN1_B, I_ENC_LN1_G, I_ENC_LN2_B, I_ENC_LN2_G,
  I_ENC_MHA_WK_B, I_ENC_MHA_WK_W, I_ENC_MHA_WO_B, I_ENC_MHA_WO_W,
  I_ENC_MHA_WQ_B, I_ENC_MHA_WQ_W, I_ENC_MHA_WV_B, I_ENC_MHA_WV_W
};
#define IN(i) ((const float*)d_in[(i)])

// ---------------- host-side helpers ----------------
static void G(hipStream_t st, int Z, int M, int N, int K,
              const float* A, int lda, long long sA1, long long sA2, int zDivA,
              const float* Bw, int ldb, long long sB1, long long sB2, int zDivB, int transB,
              float* C, int ldc, long long sC1, long long sC2, int zDivC,
              const float* bias, const float* scale, int relu) {
  GemmP p;
  p.A = A; p.Bw = Bw; p.C = C; p.bias = bias; p.scale = scale;
  p.sA1 = sA1; p.sA2 = sA2; p.sB1 = sB1; p.sB2 = sB2; p.sC1 = sC1; p.sC2 = sC2;
  p.zDivA = zDivA; p.zDivB = zDivB; p.zDivC = zDivC;
  p.lda = lda; p.ldb = ldb; p.ldc = ldc;
  p.M = M; p.N = N; p.K = K; p.transB = transB; p.relu = relu;
  dim3 grid(N / BN, M / BM, Z);
  bool fast = ((K & 31) == 0) && ((lda & 3) == 0) && ((ldb & 3) == 0) && ((M & 127) == 0);
  if (fast) k_gemm_wmma<true ><<<grid, dim3(256, 1, 1), 0, st>>>(p);
  else      k_gemm_wmma<false><<<grid, dim3(256, 1, 1), 0, st>>>(p);
}

static void mha_run(hipStream_t st, const float* qin, const float* kvin, float* outp,
                    const float* wqw, const float* wqb, const float* wkw, const float* wkb,
                    const float* wvw, const float* wvb, const float* wow, const float* wob,
                    float* tq, float* tk, float* tv, float* att, float* tctx) {
  const long long SD = (long long)SEQ * DM;
  G(st, 1, BATCH*SEQ, DM, DM, qin,  DM, 0,0,1, wqw, DM, 0,0,1, 0, tq, DM, 0,0,1, wqb, nullptr, 0);
  G(st, 1, BATCH*SEQ, DM, DM, kvin, DM, 0,0,1, wkw, DM, 0,0,1, 0, tk, DM, 0,0,1, wkb, nullptr, 0);
  G(st, 1, BATCH*SEQ, DM, DM, kvin, DM, 0,0,1, wvw, DM, 0,0,1, 0, tv, DM, 0,0,1, wvb, nullptr, 0);
  // scores[b,h] = Q[b,h] @ K[b,h]^T
  G(st, BATCH*NH, SEQ, SEQ, DHEAD,
    tq, DM, SD, DHEAD, NH,
    tk, DM, SD, DHEAD, NH, 1,
    att, SEQ, (long long)SEQ*SEQ, 0, 1,
    nullptr, nullptr, 0);
  k_softmax<<<BATCH*NH*SEQ, 256, 0, st>>>(att, 0.0883883476483184f /*1/sqrt(128)*/);
  // ctx[b,h] = P @ V[b,h]
  G(st, BATCH*NH, SEQ, DHEAD, SEQ,
    att, SEQ, (long long)SEQ*SEQ, 0, 1,
    tv, DM, SD, DHEAD, NH, 0,
    tctx, DM, SD, DHEAD, NH,
    nullptr, nullptr, 0);
  G(st, 1, BATCH*SEQ, DM, DM, tctx, DM, 0,0,1, wow, DM, 0,0,1, 0, outp, DM, 0,0,1, wob, nullptr, 0);
}

static void transformer_run(hipStream_t st, void* const* d_in,
                            const float* src, const float* tgt, float* outp,
                            float* tq, float* tk, float* tv, float* att,
                            float* t0, float* t1, float* t2, float* tff, float* encb) {
  const int ROWS = BATCH * SEQ;
  // ---- encoder on src ----
  mha_run(st, src, src, t0,
          IN(I_ENC_MHA_WQ_W), IN(I_ENC_MHA_WQ_B), IN(I_ENC_MHA_WK_W), IN(I_ENC_MHA_WK_B),
          IN(I_ENC_MHA_WV_W), IN(I_ENC_MHA_WV_B), IN(I_ENC_MHA_WO_W), IN(I_ENC_MHA_WO_B),
          tq, tk, tv, att, t2);
  k_add_ln<<<ROWS, 256, 0, st>>>(src, t0, IN(I_ENC_LN1_G), IN(I_ENC_LN1_B), t1);
  G(st, 1, ROWS, DFF2, DM, t1, DM, 0,0,1, IN(I_ENC_FFN_D1_W), DFF2, 0,0,1, 0, tff, DFF2, 0,0,1, IN(I_ENC_FFN_D1_B), nullptr, 1);
  G(st, 1, ROWS, DM, DFF2, tff, DFF2, 0,0,1, IN(I_ENC_FFN_D2_W), DM, 0,0,1, 0, t0, DM, 0,0,1, IN(I_ENC_FFN_D2_B), nullptr, 0);
  k_add_ln<<<ROWS, 256, 0, st>>>(t1, t0, IN(I_ENC_LN2_G), IN(I_ENC_LN2_B), encb);
  // ---- decoder on tgt with encb ----
  mha_run(st, tgt, tgt, t0,
          IN(I_DEC_MHA1_WQ_W), IN(I_DEC_MHA1_WQ_B), IN(I_DEC_MHA1_WK_W), IN(I_DEC_MHA1_WK_B),
          IN(I_DEC_MHA1_WV_W), IN(I_DEC_MHA1_WV_B), IN(I_DEC_MHA1_WO_W), IN(I_DEC_MHA1_WO_B),
          tq, tk, tv, att, t2);
  k_add_ln<<<ROWS, 256, 0, st>>>(tgt, t0, IN(I_DEC_LN1_G), IN(I_DEC_LN1_B), t1);
  mha_run(st, t1, encb, t0,
          IN(I_DEC_MHA2_WQ_W), IN(I_DEC_MHA2_WQ_B), IN(I_DEC_MHA2_WK_W), IN(I_DEC_MHA2_WK_B),
          IN(I_DEC_MHA2_WV_W), IN(I_DEC_MHA2_WV_B), IN(I_DEC_MHA2_WO_W), IN(I_DEC_MHA2_WO_B),
          tq, tk, tv, att, t2);
  k_add_ln<<<ROWS, 256, 0, st>>>(t1, t0, IN(I_DEC_LN2_G), IN(I_DEC_LN2_B), t2);
  G(st, 1, ROWS, DFF2, DM, t2, DM, 0,0,1, IN(I_DEC_FFN_D1_W), DFF2, 0,0,1, 0, tff, DFF2, 0,0,1, IN(I_DEC_FFN_D1_B), nullptr, 1);
  G(st, 1, ROWS, DM, DFF2, tff, DFF2, 0,0,1, IN(I_DEC_FFN_D2_W), DM, 0,0,1, 0, t0, DM, 0,0,1, IN(I_DEC_FFN_D2_B), nullptr, 0);
  k_add_ln<<<ROWS, 256, 0, st>>>(t2, t0, IN(I_DEC_LN3_G), IN(I_DEC_LN3_B), outp);
}

// ---------------- entry point ----------------
extern "C" void kernel_launch(void* const* d_in, const int* in_sizes, int n_in,
                              void* d_out, int out_size, void* d_ws, size_t ws_size,
                              hipStream_t stream) {
  (void)in_sizes; (void)n_in; (void)out_size; (void)ws_size;
  const float* X = (const float*)d_in[I_X];

  // bump allocator over workspace
  char* wp = (char*)d_ws;
  auto alloc = [&](size_t bytes) -> void* {
    void* p = (void*)wp;
    wp += (bytes + 255) & ~(size_t)255;
    return p;
  };
  const long long HKN = (long long)NCLOUD * NPTS * KNBR;
  const long long BSD = (long long)BATCH * SEQ * DM;

  int*   idxb   = (int*)  alloc(HKN * sizeof(int));
  float* hA     = (float*)alloc(HKN * 256 * sizeof(float));
  float* hB     = (float*)alloc(HKN * 128 * sizeof(float));
  float* feat   = (float*)alloc((long long)NCLOUD * NPTS * 512 * sizeof(float));
  float* emb    = (float*)alloc(2 * BSD * sizeof(float));          // [sf | tf]
  float* bnbuf  = (float*)alloc(5 * 1024 * sizeof(float));         // scale@l*1024, shift@l*1024+512
  float* tq     = (float*)alloc(BSD * sizeof(float));
  float* tk     = (float*)alloc(BSD * sizeof(float));
  float* tvb    = (float*)alloc(BSD * sizeof(float));
  float* att    = (float*)alloc((long long)BATCH * NH * SEQ * SEQ * sizeof(float));
  float* t0     = (float*)alloc(BSD * sizeof(float));
  float* t1     = (float*)alloc(BSD * sizeof(float));
  float* t2     = (float*)alloc(BSD * sizeof(float));
  float* tff    = (float*)alloc((long long)BATCH * SEQ * DFF2 * sizeof(float));
  float* encb   = (float*)alloc(BSD * sizeof(float));
  float* tro    = (float*)alloc(BSD * sizeof(float));
  float* se     = (float*)alloc(BSD * sizeof(float));
  float* te     = (float*)alloc(BSD * sizeof(float));
  float* matched= (float*)alloc((long long)BATCH * NPTS * 3 * sizeof(float));
  float* cents  = (float*)alloc(BATCH * 6 * sizeof(float));
  float* hmat   = (float*)alloc(BATCH * 9 * sizeof(float));

  // ---- fold BN into per-channel scale/shift ----
  const int bnC[5]   = {64, 64, 128, 256, 512};
  const int bnIdx[5] = {I_BN1_B, I_BN2_B, I_BN3_B, I_BN4_B, I_BN5_B};
  for (int l = 0; l < 5; ++l) {
    k_bnfold<<<(bnC[l] + 63) / 64, 64, 0, stream>>>(
        IN(bnIdx[l] + 1), IN(bnIdx[l] + 0), IN(bnIdx[l] + 2), IN(bnIdx[l] + 3),
        bnbuf + l * 1024, bnbuf + l * 1024 + 512, bnC[l]);
  }
#define BNSC(l) (bnbuf + (l) * 1024)
#define BNSH(l) (bnbuf + (l) * 1024 + 512)

  // ---- DGCNN ----
  k_topk<<<(NCLOUD * NPTS) / 256, 256, 0, stream>>>(X, idxb);
  k_h0<<<(int)(HKN / 256), 256, 0, stream>>>(X, idxb, hA);
  const long long S6 = HKN / NCLOUD * 6, S64 = HKN / NCLOUD * 64,
                  S128 = HKN / NCLOUD * 128, S256 = HKN / NCLOUD * 256;
  G(stream, NCLOUD, NPTS*KNBR,  64,   6, hA, 6,   S6,   0,1, IN(I_W1),  64, 0,0,1,0, hB,  64, S64,  0,1, BNSH(0), BNSC(0), 1);
  k_maxk<<<(int)((long long)NCLOUD*NPTS*64/256), 256, 0, stream>>>(hB, feat, 64, 0);
  G(stream, NCLOUD, NPTS*KNBR,  64,  64, hB, 64,  S64,  0,1, IN(I_W2),  64, 0,0,1,0, hA,  64, S64,  0,1, BNSH(1), BNSC(1), 1);
  k_maxk<<<(int)((long long)NCLOUD*NPTS*64/256), 256, 0, stream>>>(hA, feat, 64, 64);
  G(stream, NCLOUD, NPTS*KNBR, 128,  64, hA, 64,  S64,  0,1, IN(I_W3), 128, 0,0,1,0, hB, 128, S128, 0,1, BNSH(2), BNSC(2), 1);
  k_maxk<<<(int)((long long)NCLOUD*NPTS*128/256), 256, 0, stream>>>(hB, feat, 128, 128);
  G(stream, NCLOUD, NPTS*KNBR, 256, 128, hB, 128, S128, 0,1, IN(I_W4), 256, 0,0,1,0, hA, 256, S256, 0,1, BNSH(3), BNSC(3), 1);
  k_maxk<<<(int)((long long)NCLOUD*NPTS*256/256), 256, 0, stream>>>(hA, feat, 256, 256);
  // w5: de-interleave clouds -> emb = [sf(16,1024,512) | tf(16,1024,512)]
  G(stream, NCLOUD, NPTS, 512, 512,
    feat, 512, (long long)NPTS*512, 0, 1,
    IN(I_W5), 512, 0, 0, 1, 0,
    emb, 512, (long long)SEQ*DM, BSD, 2,
    BNSH(4), BNSC(4), 1);
  const float* sf = emb;
  const float* tf = emb + BSD;

  // ---- transformers (shared weights) ----
  transformer_run(stream, d_in, tf, sf, tro, tq, tk, tvb, att, t0, t1, t2, tff, encb);
  k_add<<<(int)((BSD + 255) / 256), 256, 0, stream>>>(sf, tro, se, BSD);
  transformer_run(stream, d_in, sf, tf, tro, tq, tk, tvb, att, t0, t1, t2, tff, encb);
  k_add<<<(int)((BSD + 255) / 256), 256, 0, stream>>>(tf, tro, te, BSD);

  // ---- pointer + SVD head ----
  G(stream, BATCH, SEQ, SEQ, DM,
    se, DM, (long long)SEQ*DM, 0, 1,
    te, DM, (long long)SEQ*DM, 0, 1, 1,
    att, SEQ, (long long)SEQ*SEQ, 0, 1,
    nullptr, nullptr, 0);
  k_softmax<<<BATCH * SEQ, 256, 0, stream>>>(att, 0.0441941738241592f /*1/sqrt(512)*/);
  k_matched<<<BATCH * SEQ, 256, 0, stream>>>(att, X, matched);
  k_centroid<<<BATCH, 256, 0, stream>>>(X, matched, cents);
  k_hmat<<<BATCH, 256, 0, stream>>>(X, matched, cents, hmat);
  k_svd<<<1, 32, 0, stream>>>(hmat, cents, (float*)d_out);
}
